// Seq2Seq_61718680043688
// MI455X (gfx1250) — matrix operand order repaired and checked by reading, not compiled, not run on hardware
//
#include <hip/hip_runtime.h>
#include <hip/hip_bf16.h>
#include <math.h>

typedef __bf16 bf16_t;
typedef __attribute__((ext_vector_type(16))) __bf16 bf16x16;
typedef __attribute__((ext_vector_type(8)))  float  floatx8;

#define BM 32
#define BN 128
#define BK 32
#define SA 36   // padded LDS A stride (bf16) -> 72B rows: 8B-aligned for async b64, bank-clean

union FragU { unsigned int u[8]; bf16x16 v; };

// ---------------- small utility kernels ----------------

__global__ void k_f32_to_bf16(const float* __restrict__ src, bf16_t* __restrict__ dst, int n) {
  int i = blockIdx.x * blockDim.x + threadIdx.x;
  if (i < n) dst[i] = (bf16_t)src[i];
}

// targets (B,64,256) f32 -> (B,63,256) bf16
__global__ void k_targets_bf16(const float* __restrict__ src, bf16_t* __restrict__ dst, int n) {
  int i = blockIdx.x * blockDim.x + threadIdx.x;
  if (i < n) {
    int b = i / (63 * 256);
    int r = i - b * (63 * 256);
    int t = r >> 8;
    int c = r & 255;
    dst[i] = (bf16_t)src[((b * 64) + t) * 256 + c];
  }
}

// transpose + convert + K-pair interleave:
// src (R,C) f32 row-major; logical Bt is (C,R); element (k=c, n=r) stored at
// dst[(k>>1)*(2R) + 2n + (k&1)]  -> each K-pair row is 2R bf16 contiguous.
__global__ void k_transpose_bf16(const float* __restrict__ src, bf16_t* __restrict__ dst,
                                 int R, int C) {
  __shared__ float tile[32][33];
  const int tx = threadIdx.x & 31;
  const int ty = threadIdx.x >> 5;       // 0..7
  const int c0 = blockIdx.x * 32;
  const int r0 = blockIdx.y * 32;
#pragma unroll
  for (int i = 0; i < 32; i += 8)
    tile[ty + i][tx] = src[(size_t)(r0 + ty + i) * C + (c0 + tx)];
  __syncthreads();
#pragma unroll
  for (int i = 0; i < 32; i += 8) {
    int k = c0 + ty + i;
    int n = r0 + tx;
    dst[(size_t)(k >> 1) * (2 * (size_t)R) + 2 * n + (k & 1)] = (bf16_t)tile[tx][ty + i];
  }
}

__global__ void k_bias_sum(const float* __restrict__ a, const float* __restrict__ b,
                           float* __restrict__ o, int n) {
  int i = blockIdx.x * blockDim.x + threadIdx.x;
  if (i < n) o[i] = a[i] + b[i];
}

__global__ void k_zero_u32(unsigned int* __restrict__ p, int n) {
  int i = blockIdx.x * blockDim.x + threadIdx.x;
  if (i < n) p[i] = 0u;
}

// zero d_out rows at t==0 for every batch
__global__ void k_zero_out_t0(float* __restrict__ out, int V) {
  int i = blockIdx.x * blockDim.x + threadIdx.x;   // 32*V threads
  int b = i / V;
  int v = i - b * V;
  out[(size_t)b * 64 * V + v] = 0.0f;
}

// ---------------- WMMA GEMM: C = A(M,K) @ Bt(K,N) [+ bias][+ addv] ----------------
// A bf16 row-major (lda); Bt bf16 in K-pair-interleaved layout (see k_transpose_bf16).
// Double-buffered LDS tiles filled with global_load_async_to_lds_* (ASYNCcnt),
// overlapping tile k+1 DMA with tile k WMMA.
template <bool ROWMAP>
__global__ __launch_bounds__(256) void k_wmma_gemm(
    const bf16_t* __restrict__ A, int lda,
    const bf16_t* __restrict__ Bt,
    const float* __restrict__ bias,
    const float* __restrict__ addv, int add_stride,
    float* __restrict__ C, int ldc,
    int N, int K) {
  __shared__ __align__(16) bf16_t lA[2][BM * SA];           // 2 x 2304 B
  __shared__ __align__(16) unsigned int lB[2][(BK / 2) * BN]; // 2 x 8192 B

  const int tid  = threadIdx.x;
  const int wave = tid >> 5;
  const int lane = tid & 31;
  const int g    = lane >> 4;     // lane half-group
  const int ln   = lane & 15;
  const int mBlock = blockIdx.y * BM;
  const int nBlock = blockIdx.x * BN;
  const int nWave  = wave * 16;   // each wave: 32 rows x 16 cols

  // LDS byte offsets (low 32 bits of flat address == addrspace(3) offset)
  const unsigned ldsA[2] = {(unsigned)(uintptr_t)&lA[0][0], (unsigned)(uintptr_t)&lA[1][0]};
  const unsigned ldsB[2] = {(unsigned)(uintptr_t)&lB[0][0], (unsigned)(uintptr_t)&lB[1][0]};

  // staging geometry: A -> 256 lanes x 8B (32 rows x 64B); B -> 2 x (8 rows x 32 lanes x 16B)
  const int ar = tid >> 3, ac = tid & 7;
  const int bw = wave, bl = lane;

  auto stage = [&](int k0, int buf) {
    unsigned la = ldsA[buf] + (unsigned)(ar * (SA * 2) + ac * 8);
    const char* ga = (const char*)A + ((size_t)(mBlock + ar) * lda + k0 + ac * 4) * 2;
    asm volatile("global_load_async_to_lds_b64 %0, %1, off" :: "v"(la), "v"(ga) : "memory");
    const size_t kp0 = (size_t)(k0 >> 1);
#pragma unroll
    for (int q = 0; q < 2; ++q) {
      int kp = bw + q * 8;
      unsigned lb = ldsB[buf] + (unsigned)(kp * (BN * 4) + bl * 16);
      const char* gb = (const char*)Bt + ((kp0 + kp) * (size_t)N + nBlock) * 4 + (size_t)bl * 16;
      asm volatile("global_load_async_to_lds_b128 %0, %1, off" :: "v"(lb), "v"(gb) : "memory");
    }
  };

  floatx8 acc0 = {};
  floatx8 acc1 = {};

  const int KT = K / BK;
  stage(0, 0);
  for (int kt = 0; kt < KT; ++kt) {
    const int buf = kt & 1;
    if (kt + 1 < KT) {
      stage((kt + 1) * BK, buf ^ 1);                       // overlap next tile DMA
      asm volatile("s_wait_asynccnt 0x3" ::: "memory");    // tile kt landed; kt+1 in flight
    } else {
      asm volatile("s_wait_asynccnt 0x0" ::: "memory");
    }
    __syncthreads();

    const bf16_t* lAb = lA[buf];
    const unsigned int* lBb = lB[buf];
    FragU a0, a1, bF;
#pragma unroll
    for (int i = 0; i < 8; ++i) {
      const int kk = ((i < 4) ? 0 : 16) + g * 8 + (i & 3) * 2;
      a0.u[i] = *(const unsigned int*)(&lAb[(ln)      * SA + kk]);
      a1.u[i] = *(const unsigned int*)(&lAb[(16 + ln) * SA + kk]);
      bF.u[i] = lBb[(((i < 4) ? 0 : 8) + g * 4 + (i & 3)) * BN + nWave + ln];
    }
    acc0 = __builtin_amdgcn_wmma_f32_16x16x32_bf16(false, a0.v, false, bF.v,
                                                   (short)0, acc0, false, false);
    acc1 = __builtin_amdgcn_wmma_f32_16x16x32_bf16(false, a1.v, false, bF.v,
                                                   (short)0, acc1, false, false);
    __syncthreads();   // all reads of buf done before it is overwritten next iter
  }

  // epilogue: C/D layout -> lane g*16+ln holds (M = r + 8*g, N = ln) in VGPR r
  const int n = nBlock + nWave + ln;
  const float bv = bias ? bias[n] : 0.0f;
#pragma unroll
  for (int r = 0; r < 8; ++r) {
    int m0 = g * 8 + r;
    int m1 = m0 + 16;
    float v0 = acc0[r] + bv;
    float v1 = acc1[r] + bv;
    if (addv) {
      v0 += addv[(size_t)(mBlock + m0) * add_stride + n];
      v1 += addv[(size_t)(mBlock + m1) * add_stride + n];
    }
    int gm0 = mBlock + m0, gm1 = mBlock + m1;
    if (ROWMAP) {
      gm0 = (gm0 / 63) * 64 + (gm0 % 63) + 1;
      gm1 = (gm1 / 63) * 64 + (gm1 % 63) + 1;
    }
    C[(size_t)gm0 * ldc + n] = v0;
    C[(size_t)gm1 * ldc + n] = v1;
  }
}

// ---------------- LSTM gate elementwise ----------------

__device__ __forceinline__ float sigmf(float x) { return 1.0f / (1.0f + __expf(-x)); }

__global__ void k_lstm_gate(const float* __restrict__ gmat,  // (32,4096)
                            float* __restrict__ c,           // (32,1024)
                            bf16_t* __restrict__ h,          // (32,1024)
                            bf16_t* __restrict__ seq,        // (32,T,1024)
                            int t, int T) {
  int i = blockIdx.x * blockDim.x + threadIdx.x;   // 32*1024
  int b = i >> 10;
  int j = i & 1023;
  const float* gr = gmat + (size_t)b * 4096;
  float gi = gr[j], gf = gr[1024 + j], gg = gr[2048 + j], go = gr[3072 + j];
  float cn = sigmf(gf) * c[i] + sigmf(gi) * tanhf(gg);
  float hn = sigmf(go) * tanhf(cn);
  c[i] = cn;
  bf16_t hb = (bf16_t)hn;
  h[i] = hb;
  seq[((size_t)b * T + t) * 1024 + j] = hb;
}

// ---------------- in-place log-softmax over V per (b,t>=1) row ----------------

__global__ __launch_bounds__(256) void k_logsoftmax(float* __restrict__ out, int V) {
  const int r = blockIdx.x;   // 0..2015
  float* p = out + (size_t)((r / 63) * 64 + (r % 63) + 1) * V;
  __shared__ float red[8];
  const int tid = threadIdx.x, lane = tid & 31, w = tid >> 5;

  float mx = -3.402823466e+38f;
  for (int i = tid; i < V; i += 256) mx = fmaxf(mx, p[i]);
#pragma unroll
  for (int off = 16; off; off >>= 1) mx = fmaxf(mx, __shfl_down(mx, off, 32));
  if (lane == 0) red[w] = mx;
  __syncthreads();
  if (tid == 0) {
    float m = red[0];
#pragma unroll
    for (int i = 1; i < 8; ++i) m = fmaxf(m, red[i]);
    red[0] = m;
  }
  __syncthreads();
  mx = red[0];
  __syncthreads();

  float s = 0.0f;
  for (int i = tid; i < V; i += 256) s += __expf(p[i] - mx);
#pragma unroll
  for (int off = 16; off; off >>= 1) s += __shfl_down(s, off, 32);
  if (lane == 0) red[w] = s;
  __syncthreads();
  if (tid == 0) {
    float t2 = 0.0f;
#pragma unroll
    for (int i = 0; i < 8; ++i) t2 += red[i];
    red[0] = t2;
  }
  __syncthreads();
  const float lse = mx + __logf(red[0]);
  for (int i = tid; i < V; i += 256) p[i] = p[i] - lse;
}

// ---------------- host orchestration ----------------

extern "C" void kernel_launch(void* const* d_in, const int* in_sizes, int n_in,
                              void* d_out, int out_size, void* d_ws, size_t ws_size,
                              hipStream_t stream) {
  (void)in_sizes; (void)n_in; (void)out_size; (void)ws_size;
  const int Bb = 32, S = 64, Td = 63, INd = 256, H = 1024, G = 4096, V = 32000;

  const float* inputs  = (const float*)d_in[0];
  const float* targets = (const float*)d_in[1];
  const float* eWih[2] = {(const float*)d_in[2],  (const float*)d_in[6]};
  const float* eWhh[2] = {(const float*)d_in[3],  (const float*)d_in[7]};
  const float* ebih[2] = {(const float*)d_in[4],  (const float*)d_in[8]};
  const float* ebhh[2] = {(const float*)d_in[5],  (const float*)d_in[9]};
  const float* dWih[2] = {(const float*)d_in[10], (const float*)d_in[14]};
  const float* dWhh[2] = {(const float*)d_in[11], (const float*)d_in[15]};
  const float* dbih[2] = {(const float*)d_in[12], (const float*)d_in[16]};
  const float* dbhh[2] = {(const float*)d_in[13], (const float*)d_in[17]};
  const float* outW = (const float*)d_in[18];
  const float* outb = (const float*)d_in[19];
  float* out = (float*)d_out;

  char* w = (char*)d_ws;
  auto alloc = [&](size_t sz) -> void* {
    void* p = (void*)w;
    w += (sz + 255) & ~(size_t)255;
    return p;
  };

  bf16_t* WihT[4];  // pair-interleaved (K, 4096): enc0, enc1, dec0, dec1
  WihT[0] = (bf16_t*)alloc((size_t)INd * G * 2);
  WihT[1] = (bf16_t*)alloc((size_t)H   * G * 2);
  WihT[2] = (bf16_t*)alloc((size_t)INd * G * 2);
  WihT[3] = (bf16_t*)alloc((size_t)H   * G * 2);
  bf16_t* WhhT[4];  // pair-interleaved (1024, 4096)
  for (int i = 0; i < 4; ++i) WhhT[i] = (bf16_t*)alloc((size_t)H * G * 2);
  bf16_t* outWT = (bf16_t*)alloc((size_t)H * V * 2);   // pair-interleaved (1024, 32000)
  float* bsum[4];
  for (int i = 0; i < 4; ++i) bsum[i] = (float*)alloc((size_t)G * 4);
  bf16_t* xin  = (bf16_t*)alloc((size_t)Bb * S  * INd * 2);
  bf16_t* tgt  = (bf16_t*)alloc((size_t)Bb * Td * INd * 2);
  bf16_t* seq0 = (bf16_t*)alloc((size_t)Bb * S  * H * 2);
  bf16_t* seq1 = (bf16_t*)alloc((size_t)Bb * S  * H * 2);
  bf16_t* dsq0 = (bf16_t*)alloc((size_t)Bb * Td * H * 2);
  bf16_t* dsq1 = (bf16_t*)alloc((size_t)Bb * Td * H * 2);
  float*  xg   = (float*)alloc((size_t)Bb * S * G * 4);
  float*  gbuf = (float*)alloc((size_t)Bb * G * 4);
  bf16_t* hbuf[2];
  float*  cbuf[2];
  for (int i = 0; i < 2; ++i) {
    hbuf[i] = (bf16_t*)alloc((size_t)Bb * H * 2);
    cbuf[i] = (float*)alloc((size_t)Bb * H * 4);
  }

  dim3 blk(256);

  // weights -> transposed + pair-interleaved bf16
  k_transpose_bf16<<<dim3(INd / 32, G / 32), blk, 0, stream>>>(eWih[0], WihT[0], G, INd);
  k_transpose_bf16<<<dim3(H   / 32, G / 32), blk, 0, stream>>>(eWih[1], WihT[1], G, H);
  k_transpose_bf16<<<dim3(INd / 32, G / 32), blk, 0, stream>>>(dWih[0], WihT[2], G, INd);
  k_transpose_bf16<<<dim3(H   / 32, G / 32), blk, 0, stream>>>(dWih[1], WihT[3], G, H);
  for (int l = 0; l < 2; ++l) {
    k_transpose_bf16<<<dim3(H / 32, G / 32), blk, 0, stream>>>(eWhh[l], WhhT[l],     G, H);
    k_transpose_bf16<<<dim3(H / 32, G / 32), blk, 0, stream>>>(dWhh[l], WhhT[2 + l], G, H);
  }
  k_transpose_bf16<<<dim3(H / 32, V / 32), blk, 0, stream>>>(outW, outWT, V, H);

  // fold bih+bhh
  k_bias_sum<<<G / 256, blk, 0, stream>>>(ebih[0], ebhh[0], bsum[0], G);
  k_bias_sum<<<G / 256, blk, 0, stream>>>(ebih[1], ebhh[1], bsum[1], G);
  k_bias_sum<<<G / 256, blk, 0, stream>>>(dbih[0], dbhh[0], bsum[2], G);
  k_bias_sum<<<G / 256, blk, 0, stream>>>(dbih[1], dbhh[1], bsum[3], G);

  // activations -> bf16
  int nin = Bb * S * INd;
  k_f32_to_bf16<<<(nin + 255) / 256, blk, 0, stream>>>(inputs, xin, nin);
  int ntg = Bb * Td * INd;
  k_targets_bf16<<<(ntg + 255) / 256, blk, 0, stream>>>(targets, tgt, ntg);

  // zero initial encoder states
  k_zero_u32<<<(Bb * H / 2 + 255) / 256, blk, 0, stream>>>((unsigned int*)hbuf[0], Bb * H / 2);
  k_zero_u32<<<(Bb * H / 2 + 255) / 256, blk, 0, stream>>>((unsigned int*)hbuf[1], Bb * H / 2);
  k_zero_u32<<<(Bb * H + 255) / 256, blk, 0, stream>>>((unsigned int*)cbuf[0], Bb * H);
  k_zero_u32<<<(Bb * H + 255) / 256, blk, 0, stream>>>((unsigned int*)cbuf[1], Bb * H);

  auto run_layer = [&](const bf16_t* X, int Kin, int T, const bf16_t* WihT_l,
                       const float* bsum_l, const bf16_t* WhhT_l,
                       bf16_t* h, float* c, bf16_t* seqout) {
    const int M = Bb * T;
    // xg = X @ WihT + (bih+bhh)
    k_wmma_gemm<false><<<dim3(G / BN, M / BM), blk, 0, stream>>>(
        X, Kin, WihT_l, bsum_l, nullptr, 0, xg, G, G, Kin);
    for (int t = 0; t < T; ++t) {
      // g = h @ WhhT + xg[:,t,:]
      k_wmma_gemm<false><<<dim3(G / BN, 1), blk, 0, stream>>>(
          h, H, WhhT_l, nullptr, xg + (size_t)t * G, T * G, gbuf, G, G, H);
      k_lstm_gate<<<(Bb * H) / 256, blk, 0, stream>>>(gbuf, c, h, seqout, t, T);
    }
  };

  // encoder (fresh zero states), decoder reuses final encoder states in-place
  run_layer(xin,  INd, S,  WihT[0], bsum[0], WhhT[0], hbuf[0], cbuf[0], seq0);
  run_layer(seq0, H,   S,  WihT[1], bsum[1], WhhT[1], hbuf[1], cbuf[1], seq1);
  run_layer(tgt,  INd, Td, WihT[2], bsum[2], WhhT[2], hbuf[0], cbuf[0], dsq0);
  run_layer(dsq0, H,   Td, WihT[3], bsum[3], WhhT[3], hbuf[1], cbuf[1], dsq1);

  // output: zeros at t=0, logits at t=1..63, then in-place log-softmax
  k_zero_out_t0<<<(Bb * V) / 256, blk, 0, stream>>>(out, V);
  k_wmma_gemm<true><<<dim3(V / BN, (Bb * Td) / BM), blk, 0, stream>>>(
      dsq1, H, outWT, outb, nullptr, 0, out, V, V, H);
  k_logsoftmax<<<Bb * Td, blk, 0, stream>>>(out, V);
}
